// GCN_24739011625365
// MI455X (gfx1250) — matrix operand order repaired and checked by reading, not compiled, not run on hardware
//
#include <hip/hip_runtime.h>
#include <hip/hip_bf16.h>

typedef __attribute__((ext_vector_type(16))) _Float16 v16h;
typedef __attribute__((ext_vector_type(8)))  float    v8f;

// ---------------------------------------------------------------------------
// Utility kernels
// ---------------------------------------------------------------------------
__global__ void cvt_f32_to_f16(const float* __restrict__ src,
                               _Float16* __restrict__ dst, int n) {
    int stride = gridDim.x * blockDim.x;
    for (int i = blockIdx.x * blockDim.x + threadIdx.x; i < n; i += stride)
        dst[i] = (_Float16)src[i];
}

__global__ void zero_f32(float* __restrict__ p, int n) {
    int stride = gridDim.x * blockDim.x;
    for (int i = blockIdx.x * blockDim.x + threadIdx.x; i < n; i += stride)
        p[i] = 0.0f;
}

__global__ void count_deg(const long long* __restrict__ col,
                          float* __restrict__ deg, int E) {
    int stride = gridDim.x * blockDim.x;
    for (int e = blockIdx.x * blockDim.x + threadIdx.x; e < E; e += stride)
        atomicAdd(&deg[(int)col[e]], 1.0f);
}

__global__ void compute_dis(const float* __restrict__ deg,
                            float* __restrict__ dis, int n) {
    int stride = gridDim.x * blockDim.x;
    for (int i = blockIdx.x * blockDim.x + threadIdx.x; i < n; i += stride) {
        float d = deg[i] + 1.0f;           // +1 self loop -> always > 0
        dis[i] = rsqrtf(d);
    }
}

// ---------------------------------------------------------------------------
// Pack W (f16, row-major [128 x N]) into WMMA B-fragment order:
//   Bp[tile t][kstep s][lane l][elem i] = B[(32s + 8*(l/16) + i + (i>=8 ? 8 : 0))*N + 16t + (l%16)]
// so the per-wave B fragment is 16 contiguous f16 (2x global_load_b128).
// ---------------------------------------------------------------------------
__global__ void pack_b_frag(const _Float16* __restrict__ B,
                            _Float16* __restrict__ Bp, int N) {
    int total  = 128 * N;                  // K = 128
    int stride = gridDim.x * blockDim.x;
    for (int idx = blockIdx.x * blockDim.x + threadIdx.x; idx < total; idx += stride) {
        int i    = idx & 15;
        int lane = (idx >> 4) & 31;
        int s    = (idx >> 9) & 3;
        int t    = idx >> 11;
        int k    = s * 32 + 8 * (lane >> 4) + i + (i >= 8 ? 8 : 0);
        int n    = t * 16 + (lane & 15);
        Bp[idx]  = B[(size_t)k * N + n];
    }
}

// ---------------------------------------------------------------------------
// WMMA GEMM: C[M x N] = A[M x 128] * B[128 x N], f16 in, f32 out.
// One wave computes a 16x16 tile; block has N/16 waves covering one 16-row strip.
// Fragment layouts per CDNA5 ISA 7.12.2 (wave32):
//   A 16x32 f16 : lane l -> row = l%16 ; elems 0..7  -> K = 8*(l/16)+i
//                                         elems 8..15 -> K = 16+8*(l/16)+(i-8)
//   B: pre-packed fragment order (see pack_b_frag)
//   C/D 16x16 f32: elem v -> row = v + 8*(l/16), col = l%16
// ---------------------------------------------------------------------------
template<int N>
__global__ __launch_bounds__(256)
void wmma_gemm_k128(const _Float16* __restrict__ A,
                    const _Float16* __restrict__ Bp,   // packed fragments
                    float* __restrict__ C, int M) {
    constexpr int K = 128;
    const int m0   = blockIdx.x * 16;
    const int wave = threadIdx.x >> 5;
    const int lane = threadIdx.x & 31;
    const int n0   = wave * 16;
    const int half = lane >> 4;   // 0 or 1
    const int l16  = lane & 15;

    int arow = m0 + l16;
    if (arow >= M) arow = M - 1;  // clamp loads; EXEC must stay all-1 for WMMA

    const _Float16* aRow   = A + (size_t)arow * K + 8 * half;
    const _Float16* bpBase = Bp + (((size_t)wave * 4) * 32 + lane) * 16;

    v8f c = {};
    #pragma unroll
    for (int s = 0; s < 4; ++s) {
        const int kk = s * 32;
        v16h a, b;
        const _Float16* ap = aRow + kk;
        #pragma unroll
        for (int i = 0; i < 8; ++i) a[i]     = ap[i];
        #pragma unroll
        for (int i = 0; i < 8; ++i) a[8 + i] = ap[16 + i];

        const _Float16* bp = bpBase + (size_t)s * 32 * 16;
        #pragma unroll
        for (int i = 0; i < 16; ++i) b[i] = bp[i];   // 2x b128, contiguous

        c = __builtin_amdgcn_wmma_f32_16x16x32_f16(
                /*neg_a=*/false, a, /*neg_b=*/false, b,
                /*c_mod=*/(short)0, c, /*reuse_a=*/false, /*reuse_b=*/false);
    }

    const int crow = m0 + 8 * half;
    float* cp = C + (size_t)crow * N + (n0 + l16);
    if (m0 + 16 <= M) {
        // full tile (every tile when M % 16 == 0): 8 unguarded coalesced stores
        #pragma unroll
        for (int v = 0; v < 8; ++v) cp[(size_t)v * N] = c[v];
    } else {
        #pragma unroll
        for (int v = 0; v < 8; ++v)
            if (crow + v < M) cp[(size_t)v * N] = c[v];
    }
}

// ---------------------------------------------------------------------------
// Self-loop init: acc[i][f] = xw[i][f] * dis[i]^2   (optionally + bias)
// ---------------------------------------------------------------------------
__global__ void init_self(const float* __restrict__ xw,
                          const float* __restrict__ dis,
                          const float* __restrict__ bias,  // may be null
                          float* __restrict__ acc,
                          int M, int F) {
    int stride = gridDim.x * blockDim.x;
    int total  = M * F;
    for (int idx = blockIdx.x * blockDim.x + threadIdx.x; idx < total; idx += stride) {
        int node = idx / F;
        int f    = idx - node * F;
        float d  = dis[node];
        float v  = xw[idx] * d * d;
        if (bias) v += bias[f];
        acc[idx] = v;
    }
}

// ---------------------------------------------------------------------------
// Edge scatter: one wave per edge, lanes stride over features.
// acc[col[e]][f] += xw[row[e]][f] * dis[row]*dis[col]
// Gathers and atomics stay L2-resident (xw/acc are 25.6 MB vs 192 MB L2).
// ---------------------------------------------------------------------------
__global__ __launch_bounds__(256)
void scatter_edges(const float* __restrict__ xw,
                   const long long* __restrict__ row,
                   const long long* __restrict__ col,
                   const float* __restrict__ dis,
                   float* __restrict__ acc,
                   int E, int F) {
    int wave   = (blockIdx.x * blockDim.x + threadIdx.x) >> 5;
    int lane   = threadIdx.x & 31;
    int nwaves = (gridDim.x * blockDim.x) >> 5;
    for (int e = wave; e < E; e += nwaves) {
        int en = e + nwaves;
        if (en < E) {                      // prefetch next edge's indices
            __builtin_prefetch(&row[en], 0, 1);   // -> global_prefetch_b8
            __builtin_prefetch(&col[en], 0, 1);
        }
        int r = (int)row[e];
        int c = (int)col[e];
        float nrm = dis[r] * dis[c];
        const float* src = xw  + (size_t)r * F;
        float*       dst = acc + (size_t)c * F;
        for (int f = lane; f < F; f += 32)
            atomicAdd(&dst[f], src[f] * nrm);
    }
}

// ---------------------------------------------------------------------------
// h = relu(acc + b) -> f16 (input to second GEMM)
// ---------------------------------------------------------------------------
__global__ void bias_relu_f16(const float* __restrict__ acc,
                              const float* __restrict__ bias,
                              _Float16* __restrict__ out,
                              int M, int F) {
    int stride = gridDim.x * blockDim.x;
    int total  = M * F;
    for (int idx = blockIdx.x * blockDim.x + threadIdx.x; idx < total; idx += stride) {
        int node = idx / F;
        int f    = idx - node * F;
        float v  = acc[idx] + bias[f];
        out[idx] = (_Float16)(v > 0.0f ? v : 0.0f);
    }
}

// ---------------------------------------------------------------------------
extern "C" void kernel_launch(void* const* d_in, const int* in_sizes, int n_in,
                              void* d_out, int out_size, void* d_ws, size_t ws_size,
                              hipStream_t stream) {
    const float*     x  = (const float*)d_in[0];
    const long long* ei = (const long long*)d_in[1];   // [2, E] int64
    const float*     W1 = (const float*)d_in[2];
    const float*     b1 = (const float*)d_in[3];
    const float*     W2 = (const float*)d_in[4];
    const float*     b2 = (const float*)d_in[5];
    float*           out = (float*)d_out;

    const int Fin = 128, H = 128, Fout = 64;
    const int M = in_sizes[0] / Fin;        // 50000
    const int E = in_sizes[1] / 2;          // 800000
    const long long* row = ei;
    const long long* col = ei + E;

    // Workspace carve-out
    char*  base = (char*)d_ws;
    size_t off  = 0;
    auto alloc = [&](size_t bytes) -> void* {
        off = (off + 255) & ~(size_t)255;
        void* p = base + off;
        off += bytes;
        return p;
    };
    _Float16* xh  = (_Float16*)alloc((size_t)M * Fin * 2);
    _Float16* W1h = (_Float16*)alloc((size_t)Fin * H * 2);
    _Float16* W2h = (_Float16*)alloc((size_t)H * Fout * 2);
    _Float16* W1p = (_Float16*)alloc((size_t)Fin * H * 2);   // packed fragments
    _Float16* W2p = (_Float16*)alloc((size_t)H * Fout * 2);
    float*    deg = (float*)alloc((size_t)M * 4);
    float*    dis = (float*)alloc((size_t)M * 4);
    float*    xw  = (float*)alloc((size_t)M * H * 4);
    float*    acc = (float*)alloc((size_t)M * H * 4);
    _Float16* hh  = (_Float16*)alloc((size_t)M * H * 2);
    float*    hw  = (float*)alloc((size_t)M * Fout * 4);
    (void)ws_size; (void)n_in; (void)out_size;

    const int TB = 256;
    const int gCvtX = ((M * Fin) + TB - 1) / TB;

    // 1. f16 conversions + fragment packing of weights
    cvt_f32_to_f16<<<gCvtX, TB, 0, stream>>>(x,  xh,  M * Fin);
    cvt_f32_to_f16<<<(Fin * H + TB - 1) / TB, TB, 0, stream>>>(W1, W1h, Fin * H);
    cvt_f32_to_f16<<<(H * Fout + TB - 1) / TB, TB, 0, stream>>>(W2, W2h, H * Fout);
    pack_b_frag<<<(Fin * H + TB - 1) / TB, TB, 0, stream>>>(W1h, W1p, H);
    pack_b_frag<<<(H * Fout + TB - 1) / TB, TB, 0, stream>>>(W2h, W2p, Fout);

    // 2. degrees + normalization
    zero_f32<<<(M + TB - 1) / TB, TB, 0, stream>>>(deg, M);
    count_deg<<<(E + TB - 1) / TB, TB, 0, stream>>>(col, deg, E);
    compute_dis<<<(M + TB - 1) / TB, TB, 0, stream>>>(deg, dis, M);

    const int mtiles = (M + 15) / 16;       // 3125
    const int gElem1 = ((M * H) + TB - 1) / TB;
    const int gScat  = 4096;                // 32768 waves grid-striding over E edges

    // 3. layer 1: xw = xh @ W1   (WMMA, 8 waves/block -> 128 cols)
    wmma_gemm_k128<128><<<mtiles, 256, 0, stream>>>(xh, W1p, xw, M);

    // 4. layer 1 aggregate
    init_self<<<gElem1, TB, 0, stream>>>(xw, dis, nullptr, acc, M, H);
    scatter_edges<<<gScat, TB, 0, stream>>>(xw, row, col, dis, acc, E, H);
    bias_relu_f16<<<gElem1, TB, 0, stream>>>(acc, b1, hh, M, H);

    // 5. layer 2: hw = hh @ W2   (WMMA, 4 waves/block -> 64 cols)
    wmma_gemm_k128<64><<<mtiles, 128, 0, stream>>>(hh, W2p, hw, M);

    // 6. layer 2 aggregate straight into d_out (init includes bias b2)
    const int gElem2 = ((M * Fout) + TB - 1) / TB;
    init_self<<<gElem2, TB, 0, stream>>>(hw, dis, b2, out, M, Fout);
    scatter_edges<<<gScat, TB, 0, stream>>>(hw, row, col, dis, out, E, Fout);
}
